// GAT_Expert_3109556322395
// MI455X (gfx1250) — matrix operand order repaired and checked by reading, not compiled, not run on hardware
//
#include <hip/hip_runtime.h>
#include <hip/hip_bf16.h>
#include <stdint.h>
#include <stddef.h>

// ---------------------------------------------------------------------------
// GAT forward (2 GATConv layers, H=2 heads, C=64) for gfx1250 (CDNA5).
// GEMMs: v_wmma_f32_16x16x32_bf16, 4 WMMAs per wave per K-step (16x64 strip),
// double-buffered LDS B panels, branch-free main loop, ds_load_b128 fragments.
// Edge softmax/aggregation: L2-resident atomics (192MB L2 holds node state).
// ---------------------------------------------------------------------------

typedef __attribute__((ext_vector_type(16))) __bf16   v16bf;
typedef __attribute__((ext_vector_type(2)))  __bf16   v2bf;
typedef __attribute__((ext_vector_type(8)))  float    v8f;
typedef __attribute__((ext_vector_type(4)))  unsigned v4u;

__device__ __forceinline__ unsigned f2bf_u(float f) {
  unsigned u = __float_as_uint(f);
  u += 0x7fffu + ((u >> 16) & 1u);   // round-to-nearest-even bf16
  return u >> 16;
}
__device__ __forceinline__ unsigned pkbf(float lo, float hi) {
#if __has_builtin(__builtin_amdgcn_cvt_pk_bf16_f32)
  union { v2bf v; unsigned u; } r;
  r.v = __builtin_amdgcn_cvt_pk_bf16_f32(lo, hi);
  return r.u;
#else
  return f2bf_u(lo) | (f2bf_u(hi) << 16);
#endif
}
// monotone float->uint mapping so atomicMax(uint) == float max
__device__ __forceinline__ unsigned ordkey(float f) {
  unsigned u = __float_as_uint(f);
  return (u & 0x80000000u) ? ~u : (u | 0x80000000u);
}
__device__ __forceinline__ float ordval(unsigned u) {
  return (u & 0x80000000u) ? __uint_as_float(u & 0x7fffffffu)
                           : __uint_as_float(~u);
}
__device__ __forceinline__ float leaky(float v) {
  return v > 0.f ? v : 0.2f * v;
}

// ---------------------------------------------------------------------------
// B panel staging: 32(K) x 64(N) f32 -> LDS as packed bf16 pairs.
// LDS layout: chunk(col,half) = col*2+half, 12-dword stride (48B: keeps the
// two ds_load_b128 per fragment 16B-aligned, spreads read banks).
// chunk slot p holds K-pair kp = (p<4 ? half*4+p : 8+half*4+p-4) -> dword = pk(B[2kp], B[2kp+1]).
// ---------------------------------------------------------------------------
#define BCHUNK 12
#define BPANEL (128 * BCHUNK)   // dwords per panel buffer

struct BStage { float2 g[4]; };

__device__ __forceinline__ BStage load_bstage(const float* __restrict__ B, int ldb,
                                              int k0, int colBase, int K, int Nc,
                                              int tid) {
  BStage st;
  if (k0 + 32 <= K) {            // uniform fast path (cols guaranteed by launch)
#pragma unroll
    for (int j = 0; j < 4; ++j) {
      int idx = tid + j * 256;
      int kp = idx >> 6, n = idx & 63;
      const float* bp = B + (size_t)(k0 + 2 * kp) * ldb + colBase + n;
      st.g[j].x = bp[0];
      st.g[j].y = bp[ldb];
    }
  } else {                       // K tail: zero-fill
#pragma unroll
    for (int j = 0; j < 4; ++j) {
      int idx = tid + j * 256;
      int kp = idx >> 6, n = idx & 63;
      int kk = k0 + 2 * kp, nn = colBase + n;
      st.g[j].x = (kk     < K && nn < Nc) ? B[(size_t)kk * ldb + nn]       : 0.f;
      st.g[j].y = (kk + 1 < K && nn < Nc) ? B[(size_t)(kk + 1) * ldb + nn] : 0.f;
    }
  }
  return st;
}
__device__ __forceinline__ void store_bstage(unsigned* __restrict__ dst,
                                             const BStage& st, int tid) {
#pragma unroll
  for (int j = 0; j < 4; ++j) {
    int idx = tid + j * 256;
    int kp = idx >> 6, n = idx & 63;
    dst[(n * 2 + ((kp >> 2) & 1)) * BCHUNK + ((kp & 3) | ((kp >> 3) << 2))] = pkbf(st.g[j].x, st.g[j].y);
  }
}

// A fragment (16-bit A 16x32 layout): elem i -> K = k0 + (i<8?i:i+8) + half*8
__device__ __forceinline__ void load_afrag(unsigned* __restrict__ u,
                                           const float* __restrict__ ap,
                                           int k0, int K, int half) {
  if (k0 + 32 <= K) {
    const float* p = ap + k0 + half * 8;
    float4 a0 = *(const float4*)(p);
    float4 a1 = *(const float4*)(p + 4);
    float4 a2 = *(const float4*)(p + 16);
    float4 a3 = *(const float4*)(p + 20);
    u[0] = pkbf(a0.x, a0.y); u[1] = pkbf(a0.z, a0.w);
    u[2] = pkbf(a1.x, a1.y); u[3] = pkbf(a1.z, a1.w);
    u[4] = pkbf(a2.x, a2.y); u[5] = pkbf(a2.z, a2.w);
    u[6] = pkbf(a3.x, a3.y); u[7] = pkbf(a3.z, a3.w);
  } else {
    const int kb0 = k0 + half * 8;
    const int kb1 = k0 + 16 + half * 8;
#pragma unroll
    for (int j = 0; j < 4; ++j) {
      int ka = kb0 + 2 * j, kb = kb1 + 2 * j;
      float v0 = (ka < K)     ? ap[ka]     : 0.f;
      float v1 = (ka + 1 < K) ? ap[ka + 1] : 0.f;
      float v2 = (kb < K)     ? ap[kb]     : 0.f;
      float v3 = (kb + 1 < K) ? ap[kb + 1] : 0.f;
      u[j]     = pkbf(v0, v1);
      u[4 + j] = pkbf(v2, v3);
    }
  }
}

// ---------------------------------------------------------------------------
// C[M,NcStore] = A[M,K](f32,lda) * B[K,Nc](f32,ldb) (+bias,+relu), bf16 WMMA.
// Block = 256 thr = 8 waves; tile 128 rows x 64 cols; wave = 16x64 strip.
// Software-pipelined: global loads for panel s+1 issued before the WMMAs of
// panel s, stored to the other LDS buffer after; one barrier per step.
// All 8 fragment ds_load_b128 issued before the first WMMA so the dscnt
// waits stagger (later fragments load under earlier WMMAs).
// ---------------------------------------------------------------------------
__global__ void gat_gemm_bf16_wmma(const float* __restrict__ A, int lda,
                                   const float* __restrict__ B, int ldb,
                                   const float* __restrict__ bias,
                                   float* __restrict__ C, int ldc,
                                   int M, int Nc, int NcStore, int K,
                                   int do_relu)
{
  __shared__ unsigned Bsh[2][BPANEL];
  const int tid  = threadIdx.x;
  const int lane = tid & 31;
  const int half = lane >> 4;          // 0: lanes 0-15, 1: lanes 16-31
  const int l16  = lane & 15;
  const int wave = tid >> 5;
  const int rowBase = blockIdx.x * 128 + wave * 16;
  const int colBase = blockIdx.y * 64;
  const int arow = rowBase + l16;
  const int arowC = (arow < M) ? arow : (M - 1);       // clamp: masked at store
  const float* ap = A + (size_t)arowC * lda;

  v8f acc[4] = {{}, {}, {}, {}};
  const int nsteps = (K + 31) >> 5;

  // prologue: panel 0 into buffer 0
  {
    BStage st0 = load_bstage(B, ldb, 0, colBase, K, Nc, tid);
    store_bstage(Bsh[0], st0, tid);
  }
  __syncthreads();

  for (int s = 0; s < nsteps; ++s) {
    const int cur = s & 1;
    const bool have_nx = (s + 1 < nsteps);
    BStage nx;
    if (have_nx)                       // issue next panel's global loads now
      nx = load_bstage(B, ldb, (s + 1) * 32, colBase, K, Nc, tid);

    union { v16bf v; unsigned u32[8]; } afrag;
    load_afrag(afrag.u32, ap, s * 32, K, half);
    if ((s + 2) * 32 <= K)
      __builtin_prefetch((const void*)(ap + (s + 2) * 32 + half * 8), 0, 3);

    const unsigned* bs = Bsh[cur];
    union BF { v16bf v; v4u q[2]; } bf[4];
#pragma unroll
    for (int t = 0; t < 4; ++t) {      // all 8 ds_load_b128 before first WMMA
      int base = ((t * 16 + l16) * 2 + half) * BCHUNK;
      __builtin_assume((base & 3) == 0);               // 16B-aligned chunks
      bf[t].q[0] = *(const v4u*)(bs + base);
      bf[t].q[1] = *(const v4u*)(bs + base + 4);
    }
#pragma unroll
    for (int t = 0; t < 4; ++t)
      acc[t] = __builtin_amdgcn_wmma_f32_16x16x32_bf16(
          false, afrag.v, false, bf[t].v, (short)0, acc[t], false, false);

    if (have_nx)                       // LDS store after WMMAs hide the latency
      store_bstage(Bsh[cur ^ 1], nx, tid);
    __syncthreads();
  }

  // epilogue: vgpr r, lane -> M = rowBase + r + half*8, N = colBase + t*16+l16
#pragma unroll
  for (int t = 0; t < 4; ++t) {
    int cn = colBase + t * 16 + l16;
    if (cn < NcStore) {
      float badd = bias ? bias[cn] : 0.f;
#pragma unroll
      for (int r = 0; r < 8; ++r) {
        int cm = rowBase + r + half * 8;
        if (cm < M) {
          float v = acc[t][r] + badd;
          if (do_relu) v = v > 0.f ? v : 0.f;
          C[(size_t)cm * ldc + cn] = v;
        }
      }
    }
  }
}

// ---------------------------------------------------------------------------
// a_src[n,h] = sum_c h[n,h*64+c]*att_src[h*64+c]  (same for a_dst).
// One wave per node; butterfly-reduce each 16-lane half (= one head).
// ---------------------------------------------------------------------------
__global__ void gat_att_dots(const float* __restrict__ h,
                             const float* __restrict__ att_src,
                             const float* __restrict__ att_dst,
                             float* __restrict__ a_src,
                             float* __restrict__ a_dst, int Nn)
{
  const int lane = threadIdx.x & 31;
  int gw = (blockIdx.x * blockDim.x + threadIdx.x) >> 5;
  int nW = (gridDim.x * blockDim.x) >> 5;
  const float4 as = ((const float4*)att_src)[lane];
  const float4 ad = ((const float4*)att_dst)[lane];
  for (int n = gw; n < Nn; n += nW) {
    const float4 hv = ((const float4*)(h + (size_t)n * 128))[lane];
    float ps = hv.x * as.x + hv.y * as.y + hv.z * as.z + hv.w * as.w;
    float pd = hv.x * ad.x + hv.y * ad.y + hv.z * ad.z + hv.w * ad.w;
#pragma unroll
    for (int m = 1; m < 16; m <<= 1) {
      ps += __shfl_xor(ps, m, 32);
      pd += __shfl_xor(pd, m, 32);
    }
    if ((lane & 15) == 0) {
      int head = lane >> 4;
      a_src[n * 2 + head] = ps;
      a_dst[n * 2 + head] = pd;
    }
  }
}

__global__ void gat_fill_u32(unsigned* __restrict__ p, unsigned v, long count) {
  long stride = (long)gridDim.x * blockDim.x;
  for (long i = blockIdx.x * (long)blockDim.x + threadIdx.x; i < count; i += stride)
    p[i] = v;
}

// pass 1: segment max of leaky(a_src[src]+a_dst[dst]) over dst
__global__ void gat_edge_max(const int* __restrict__ ei, int E, int Nn,
                             const float* __restrict__ a_src,
                             const float* __restrict__ a_dst,
                             unsigned* __restrict__ emax)
{
  const int tE = E + Nn;
  int stride = gridDim.x * blockDim.x;
  for (int e = blockIdx.x * blockDim.x + threadIdx.x; e < tE; e += stride) {
    int s = (e < E) ? ei[e]     : (e - E);
    int d = (e < E) ? ei[E + e] : (e - E);
#pragma unroll
    for (int hh = 0; hh < 2; ++hh) {
      float v = leaky(a_src[s * 2 + hh] + a_dst[d * 2 + hh]);
      atomicMax(&emax[d * 2 + hh], ordkey(v));
    }
  }
}

// pass 2: denom[dst] = sum exp(e - max)
__global__ void gat_edge_denom(const int* __restrict__ ei, int E, int Nn,
                               const float* __restrict__ a_src,
                               const float* __restrict__ a_dst,
                               const unsigned* __restrict__ emax,
                               float* __restrict__ denom)
{
  const int tE = E + Nn;
  int stride = gridDim.x * blockDim.x;
  for (int e = blockIdx.x * blockDim.x + threadIdx.x; e < tE; e += stride) {
    int s = (e < E) ? ei[e]     : (e - E);
    int d = (e < E) ? ei[E + e] : (e - E);
#pragma unroll
    for (int hh = 0; hh < 2; ++hh) {
      float v = leaky(a_src[s * 2 + hh] + a_dst[d * 2 + hh]);
      float m = ordval(emax[d * 2 + hh]);
      atomicAdd(&denom[d * 2 + hh], __expf(v - m));
    }
  }
}

// pass 3: agg[dst] += alpha * h[src]   (wave per edge, lane = 4 channels)
__global__ void gat_edge_agg(const int* __restrict__ ei, int E, int Nn,
                             const float* __restrict__ a_src,
                             const float* __restrict__ a_dst,
                             const unsigned* __restrict__ emax,
                             const float* __restrict__ denom,
                             const float* __restrict__ h,
                             float* __restrict__ agg)
{
  const int lane = threadIdx.x & 31;
  const int head = lane >> 4;                 // lanes 0-15: ch 0-63 (head 0)
  int gw = (blockIdx.x * blockDim.x + threadIdx.x) >> 5;
  int nW = (gridDim.x * blockDim.x) >> 5;
  const int tE = E + Nn;
  for (int e = gw; e < tE; e += nW) {
    int s = (e < E) ? ei[e]     : (e - E);
    int d = (e < E) ? ei[E + e] : (e - E);
    float v = leaky(a_src[s * 2 + head] + a_dst[d * 2 + head]);
    float m = ordval(emax[d * 2 + head]);
    float alpha = __expf(v - m) / (denom[d * 2 + head] + 1e-16f);
    float4 hv = ((const float4*)(h + (size_t)s * 128))[lane];
    float* op = agg + (size_t)d * 128 + lane * 4;
    atomicAdd(op + 0, hv.x * alpha);
    atomicAdd(op + 1, hv.y * alpha);
    atomicAdd(op + 2, hv.z * alpha);
    atomicAdd(op + 3, hv.w * alpha);
  }
}

// x_out[n, c] (ldo) = agg[n*128+c] + bias[c]  (optional relu)
__global__ void gat_bias_act(const float* __restrict__ agg,
                             const float* __restrict__ bias,
                             float* __restrict__ xo, int ldo, int Nn, int relu)
{
  long total = (long)Nn * 128;
  long stride = (long)gridDim.x * blockDim.x;
  for (long i = blockIdx.x * (long)blockDim.x + threadIdx.x; i < total; i += stride) {
    int n = (int)(i >> 7), c = (int)(i & 127);
    float v = agg[i] + bias[c];
    if (relu) v = v > 0.f ? v : 0.f;
    xo[(size_t)n * ldo + c] = v;
  }
}

// x0[n, 0..39] = logits[n, 0..39]
__global__ void gat_copy_logits(const float* __restrict__ lg,
                                float* __restrict__ x0, int Nn)
{
  long total = (long)Nn * 40;
  long stride = (long)gridDim.x * blockDim.x;
  for (long i = blockIdx.x * (long)blockDim.x + threadIdx.x; i < total; i += stride) {
    int n = (int)(i / 40), c = (int)(i % 40);
    x0[(size_t)n * 168 + c] = lg[i];
  }
}

// zero-pad W_out [128,40] -> [128,64] so the head GEMM takes the fast path
__global__ void gat_pad_wout(const float* __restrict__ W,
                             float* __restrict__ Wp)
{
  int i = blockIdx.x * blockDim.x + threadIdx.x;
  if (i < 128 * 64) {
    int k = i >> 6, n = i & 63;
    Wp[i] = (n < 40) ? W[k * 40 + n] : 0.f;
  }
}

// ---------------------------------------------------------------------------
extern "C" void kernel_launch(void* const* d_in, const int* in_sizes, int n_in,
                              void* d_out, int out_size, void* d_ws, size_t ws_size,
                              hipStream_t stream)
{
  const float* logits   = (const float*)d_in[0];
  const float* features = (const float*)d_in[1];
  const int*   ei       = (const int*)  d_in[2];
  const float* W_feat   = (const float*)d_in[3];
  const float* b_feat   = (const float*)d_in[4];
  const float* W0       = (const float*)d_in[5];
  const float* as0      = (const float*)d_in[6];
  const float* ad0      = (const float*)d_in[7];
  const float* bias0    = (const float*)d_in[8];
  const float* W1       = (const float*)d_in[9];
  const float* as1      = (const float*)d_in[10];
  const float* ad1      = (const float*)d_in[11];
  const float* bias1    = (const float*)d_in[12];
  const float* W_out    = (const float*)d_in[13];
  const float* b_out    = (const float*)d_in[14];
  float* out = (float*)d_out;

  const int Nn = in_sizes[0] / 40;
  const int E  = in_sizes[2] / 2;
  const int tE = E + Nn;

  // workspace carve-up (all 256B aligned)
  char* ws = (char*)d_ws;
  size_t off = 0;
  auto carve = [&](size_t bytes) -> char* {
    char* p = ws + off;
    off += (bytes + 255) & ~(size_t)255;
    return p;
  };
  float*    x     = (float*)   carve((size_t)Nn * 168 * 4); // x0 [N,168]; later x1/x2 [N,128]
  float*    h     = (float*)   carve((size_t)Nn * 128 * 4);
  float*    agg   = (float*)   carve((size_t)Nn * 128 * 4);
  float*    a_src = (float*)   carve((size_t)Nn * 2 * 4);
  float*    a_dst = (float*)   carve((size_t)Nn * 2 * 4);
  float*    denom = (float*)   carve((size_t)Nn * 2 * 4);
  unsigned* emax  = (unsigned*)carve((size_t)Nn * 2 * 4);
  float*    Wpad  = (float*)   carve((size_t)128 * 64 * 4);
  (void)ws_size; (void)n_in; (void)out_size;

  const int mB = (Nn + 127) / 128;                 // gemm row-tiles
  const dim3 blk256(256);
  const int edgeBlks = (tE + 255) / 256;
  const int aggBlks  = (tE + 7) / 8;               // wave per edge
  const int elemBlks = (int)(((long)Nn * 128 + 255) / 256);
  const int waveBlks = (Nn + 7) / 8;               // att dots: 8 nodes/block
  const unsigned NEG_INF_KEY = 0x007fffffu;        // ordkey(-inf)

  // ---- input assembly: x = [logits | features @ W_feat + b_feat] ----
  gat_copy_logits<<<(int)(((long)Nn * 40 + 255) / 256), blk256, 0, stream>>>(logits, x, Nn);
  gat_pad_wout<<<32, blk256, 0, stream>>>(W_out, Wpad);
  gat_gemm_bf16_wmma<<<dim3(mB, 2), blk256, 0, stream>>>(
      features, 256, W_feat, 128, b_feat, x + 40, 168, Nn, 128, 128, 256, 0);

  // ---- layer 0 ----
  gat_gemm_bf16_wmma<<<dim3(mB, 2), blk256, 0, stream>>>(
      x, 168, W0, 128, nullptr, h, 128, Nn, 128, 128, 168, 0);
  gat_att_dots<<<waveBlks, blk256, 0, stream>>>(h, as0, ad0, a_src, a_dst, Nn);
  gat_fill_u32<<<(2 * Nn + 255) / 256, blk256, 0, stream>>>(emax, NEG_INF_KEY, 2L * Nn);
  gat_fill_u32<<<(2 * Nn + 255) / 256, blk256, 0, stream>>>((unsigned*)denom, 0u, 2L * Nn);
  gat_fill_u32<<<elemBlks, blk256, 0, stream>>>((unsigned*)agg, 0u, (long)Nn * 128);
  gat_edge_max  <<<edgeBlks, blk256, 0, stream>>>(ei, E, Nn, a_src, a_dst, emax);
  gat_edge_denom<<<edgeBlks, blk256, 0, stream>>>(ei, E, Nn, a_src, a_dst, emax, denom);
  gat_edge_agg  <<<aggBlks,  blk256, 0, stream>>>(ei, E, Nn, a_src, a_dst, emax, denom, h, agg);
  gat_bias_act  <<<elemBlks, blk256, 0, stream>>>(agg, bias0, x, 128, Nn, 1); // relu -> x1

  // ---- layer 1 ----
  gat_gemm_bf16_wmma<<<dim3(mB, 2), blk256, 0, stream>>>(
      x, 128, W1, 128, nullptr, h, 128, Nn, 128, 128, 128, 0);
  gat_att_dots<<<waveBlks, blk256, 0, stream>>>(h, as1, ad1, a_src, a_dst, Nn);
  gat_fill_u32<<<(2 * Nn + 255) / 256, blk256, 0, stream>>>(emax, NEG_INF_KEY, 2L * Nn);
  gat_fill_u32<<<(2 * Nn + 255) / 256, blk256, 0, stream>>>((unsigned*)denom, 0u, 2L * Nn);
  gat_fill_u32<<<elemBlks, blk256, 0, stream>>>((unsigned*)agg, 0u, (long)Nn * 128);
  gat_edge_max  <<<edgeBlks, blk256, 0, stream>>>(ei, E, Nn, a_src, a_dst, emax);
  gat_edge_denom<<<edgeBlks, blk256, 0, stream>>>(ei, E, Nn, a_src, a_dst, emax, denom);
  gat_edge_agg  <<<aggBlks,  blk256, 0, stream>>>(ei, E, Nn, a_src, a_dst, emax, denom, h, agg);
  gat_bias_act  <<<elemBlks, blk256, 0, stream>>>(agg, bias1, x, 128, Nn, 0); // x2 (no relu)

  // ---- output head: out = x2 @ Wpad(64 cols, zero-padded) + b_out ----
  gat_gemm_bf16_wmma<<<dim3(mB, 1), blk256, 0, stream>>>(
      x, 128, Wpad, 64, b_out, out, 40, Nn, 64, 40, 128, 0);
}
